// GraphConv_33268816675169
// MI455X (gfx1250) — matrix operand order repaired and checked by reading, not compiled, not run on hardware
//
#include <hip/hip_runtime.h>

typedef _Float16 v16h __attribute__((ext_vector_type(16)));
typedef _Float16 v8h  __attribute__((ext_vector_type(8)));
typedef float    v8f  __attribute__((ext_vector_type(8)));

#define TB 8            // t-values per workgroup tile
#define C  64
#define V  25
#define VP 32           // padded V / W dim
#define K3 3
#define O  64
#define KC 192          // K3*C
#define T  512
#define NB 32
#define TV  12800       // T*V
#define NTV 409600      // NB*T*V per channel

// ---- WMMA operand loaders from LDS (wave32, v_wmma_f32_16x16x32_f16) ----
// A-matrix 16x32 f16: lane m=lane&15, h=lane>>4; elem j -> K = (j>>3)*16 + h*8 + (j&7)
// (two contiguous 16B chunks per lane -> 2x ds_load_b128)
__device__ __forceinline__ v16h lds_A_tile(const _Float16* base, int stride) {
  const int lane = threadIdx.x & 31;
  const int m = lane & 15, h = lane >> 4;
  v16h a;
#pragma unroll
  for (int j = 0; j < 16; ++j) {
    int k = ((j >> 3) << 4) + (h << 3) + (j & 7);
    a[j] = base[m * stride + k];
  }
  return a;
}
// B-matrix 32x16 f16 from N-major storage: element (K,N) at base[N*stride + K].
// lane n=lane&15, h=lane>>4; elem j -> K = (j>>3)*16 + h*8 + (j&7)  => two 16B chunks per lane.
__device__ __forceinline__ v16h lds_B_tileT(const _Float16* base, int stride) {
  const int lane = threadIdx.x & 31;
  const int n = lane & 15, h = lane >> 4;
  const _Float16* p = base + n * stride + (h << 3);
  v16h b;
#pragma unroll
  for (int j = 0; j < 16; ++j) b[j] = p[((j >> 3) << 4) + (j & 7)];
  return b;
}

// ---------------- Kernel 1: fused graph-agg + 1x1 conv (WMMA) ----------------
__global__ __launch_bounds__(128) void k1_fused_gemm(
    const float* __restrict__ x, const float* __restrict__ A,
    const float* __restrict__ W, const float* __restrict__ b,
    float* __restrict__ Y) {
  __shared__ __align__(16) _Float16 Xs[C * TB * VP];    // [c][t][w]   32 KB
  __shared__ __align__(16) _Float16 As[K3 * VP * VP];   // [k][v][w]    6 KB (N-major for B)
  __shared__ __align__(16) _Float16 Ws[O * KC];         // [o][kc]     24 KB
  __shared__ __align__(16) _Float16 Ss[TB * VP * KC];   // [t][v][kc]  96 KB (N-major for B)

  const int tid  = threadIdx.x;
  const int wave = tid >> 5;
  const int lane = tid & 31;
  const int n  = blockIdx.x >> 6;            // T/TB = 64 t-blocks
  const int t0 = (blockIdx.x & 63) * TB;

  // ---- stage inputs into LDS (f32 -> f16, zero-pad w,v to 32) ----
  for (int idx = tid; idx < C * TB * VP; idx += 128) {
    int c = idx >> 8;                 // TB*VP = 256
    int t = (idx >> 5) & (TB - 1);
    int w = idx & 31;
    float v = (w < V) ? x[((n * C + c) * T + (t0 + t)) * V + w] : 0.f;
    Xs[idx] = (_Float16)v;
  }
  for (int idx = tid; idx < K3 * VP * VP; idx += 128) {
    int k = idx >> 10;
    int v = (idx >> 5) & 31;
    int w = idx & 31;
    float a = (w < V && v < V) ? A[(k * V + v) * V + w] : 0.f;  // As[k][v][w] = A natural
    As[idx] = (_Float16)a;
  }
  for (int idx = tid; idx < O * KC; idx += 128) {
    Ws[idx] = (_Float16)W[idx];
  }
  __syncthreads();

  const int h = lane >> 4, nn = lane & 15;

  // ---- stage 1: S[t][v][(k,c)] = sum_w Xs[c][t][w] * A[k][v][w] ----
  // per (k,t): M=c(4x16) N=v(2x16) K=w(32)
  for (int p = wave; p < K3 * TB; p += 4) {
    const int k = p >> 3;
    const int t = p & (TB - 1);
    const _Float16* xbase = &Xs[t * VP];                 // row stride (c) = TB*VP
    v16h b0 = lds_B_tileT(&As[k * VP * VP], VP);         // v rows 0..15
    v16h b1 = lds_B_tileT(&As[k * VP * VP + 16 * VP], VP); // v rows 16..31 (zero pad)
#pragma unroll
    for (int mt = 0; mt < 4; ++mt) {
      v16h am = lds_A_tile(xbase + (mt * 16) * (TB * VP), TB * VP);
      v8f acc0 = {}, acc1 = {};
      acc0 = __builtin_amdgcn_wmma_f32_16x16x32_f16(false, am, false, b0,
                                                    (short)0, acc0, false, false);
      acc1 = __builtin_amdgcn_wmma_f32_16x16x32_f16(false, am, false, b1,
                                                    (short)0, acc1, false, false);
      // lane holds rows c = mt*16 + r + h*8 (8 consecutive kc at fixed v) -> one b128 store
      const int kcb = k * C + mt * 16 + h * 8;
      v8h pk0, pk1;
#pragma unroll
      for (int r = 0; r < 8; ++r) { pk0[r] = (_Float16)acc0[r]; pk1[r] = (_Float16)acc1[r]; }
      *(v8h*)&Ss[(t * VP + nn) * KC + kcb]        = pk0;
      *(v8h*)&Ss[(t * VP + 16 + nn) * KC + kcb]   = pk1;
    }
  }
  __syncthreads();

  // ---- stage 2: y[o][v] = sum_kc W[o][kc] * S[t][v][kc] + b[o] ----
  const int otile = wave;
  v16h wa[6];
#pragma unroll
  for (int kk = 0; kk < 6; ++kk)
    wa[kk] = lds_A_tile(&Ws[(otile * 16) * KC + kk * 32], KC);

  float bias[8];
#pragma unroll
  for (int r = 0; r < 8; ++r) bias[r] = b[otile * 16 + r + h * 8];

  for (int t = 0; t < TB; ++t) {
#pragma unroll
    for (int nt = 0; nt < 2; ++nt) {
      v8f acc = {};
#pragma unroll
      for (int kk = 0; kk < 6; ++kk) {
        v16h bm = lds_B_tileT(&Ss[(t * VP + nt * 16) * KC + kk * 32], KC);
        acc = __builtin_amdgcn_wmma_f32_16x16x32_f16(false, wa[kk], false, bm,
                                                     (short)0, acc, false, false);
      }
      const int vv = nt * 16 + nn;
      if (vv < V) {
#pragma unroll
        for (int r = 0; r < 8; ++r) {
          int o = otile * 16 + r + h * 8;
          Y[((size_t)(n * O + o) * T + (t0 + t)) * V + vv] = acc[r] + bias[r];
        }
      }
    }
  }
}

// ---------------- Kernel 2: per-channel batch stats -> scale/shift ----------------
__global__ __launch_bounds__(256) void k2_stats(
    const float* __restrict__ Y, const float* __restrict__ gamma,
    const float* __restrict__ beta, float* __restrict__ ss) {
  __shared__ float red[512];
  const int o = blockIdx.x;
  const int tid = threadIdx.x;
  float s = 0.f, s2 = 0.f;
  for (int n = 0; n < NB; ++n) {
    const float4* base = (const float4*)(Y + (size_t)(n * O + o) * TV);
    for (int i = tid; i < TV / 4; i += 256) {
      float4 v = base[i];
      s  += v.x + v.y + v.z + v.w;
      s2 += v.x * v.x + v.y * v.y + v.z * v.z + v.w * v.w;
    }
  }
  red[tid] = s; red[tid + 256] = s2;
  __syncthreads();
  for (int off = 128; off > 0; off >>= 1) {
    if (tid < off) { red[tid] += red[tid + off]; red[tid + 256] += red[tid + 256 + off]; }
    __syncthreads();
  }
  if (tid == 0) {
    float mean = red[0] / (float)NTV;
    float var  = red[256] / (float)NTV - mean * mean;
    float rstd = rsqrtf(var + 1e-5f);
    float sc = gamma[o] * rstd;
    ss[o]     = sc;
    ss[O + o] = beta[o] - mean * sc;
  }
}

// ---------------- Kernel 3: normalize + ReLU (float4) ----------------
__global__ __launch_bounds__(256) void k3_norm_relu(
    const float* __restrict__ Y, const float* __restrict__ ss,
    float* __restrict__ out, int total4) {
  __shared__ float sc[O], sh[O];
  for (int i = threadIdx.x; i < O; i += 256) { sc[i] = ss[i]; sh[i] = ss[O + i]; }
  __syncthreads();
  const float4* Y4 = (const float4*)Y;
  float4* out4 = (float4*)out;
  for (int idx = blockIdx.x * 256 + threadIdx.x; idx < total4; idx += gridDim.x * 256) {
    int o = ((idx << 2) / TV) & (O - 1);   // TV % 4 == 0 -> float4 never crosses channels
    float4 y = Y4[idx];
    float a = sc[o], c = sh[o];
    float4 r;
    r.x = fmaxf(y.x * a + c, 0.f);
    r.y = fmaxf(y.y * a + c, 0.f);
    r.z = fmaxf(y.z * a + c, 0.f);
    r.w = fmaxf(y.w * a + c, 0.f);
    out4[idx] = r;
  }
}

extern "C" void kernel_launch(void* const* d_in, const int* in_sizes, int n_in,
                              void* d_out, int out_size, void* d_ws, size_t ws_size,
                              hipStream_t stream) {
  const float* x     = (const float*)d_in[0];
  const float* A     = (const float*)d_in[1];
  const float* W     = (const float*)d_in[2];
  const float* b     = (const float*)d_in[3];
  const float* gamma = (const float*)d_in[4];
  const float* beta  = (const float*)d_in[5];
  float* out = (float*)d_out;

  float* Y  = (float*)d_ws;                    // N*O*T*V f32 intermediate (L2-resident)
  float* ss = Y + (size_t)NB * O * TV;         // 2*O floats: scale, shift

  k1_fused_gemm<<<NB * (T / TB), 128, 0, stream>>>(x, A, W, b, Y);
  k2_stats<<<O, 256, 0, stream>>>(Y, gamma, beta, ss);
  k3_norm_relu<<<4096, 256, 0, stream>>>(Y, ss, out, (NB * O * TV) / 4);
}
